// ISA_2336462209656
// MI455X (gfx1250) — matrix operand order repaired and checked by reading, not compiled, not run on hardware
//
#include <hip/hip_runtime.h>

#define SLOTS 8
#define DSN   256
#define NB    2048
#define BB    4
#define DIN   768
#define MENC  (BB*NB)            // 8192
#define RTOT  (BB*SLOTS*NB)      // 65536

typedef __bf16 bf16;
typedef __attribute__((ext_vector_type(16))) __bf16 v16bf;
typedef __attribute__((ext_vector_type(8)))  __bf16 v8bf;
typedef __attribute__((ext_vector_type(8)))  float  v8f;

// ---- gfx1250 async global->LDS staging (guarded; falls back to reg copy) ----
#if defined(__has_builtin)
#if __has_builtin(__builtin_amdgcn_global_load_async_to_lds_b128)
#define HAVE_ASYNC 1
#endif
#endif

typedef int v4i_t __attribute__((vector_size(16)));
typedef __attribute__((address_space(1))) v4i_t g_v4i;   // global
typedef __attribute__((address_space(3))) v4i_t l_v4i;   // LDS

#ifdef HAVE_ASYNC
#define CP16(ldsp, gp) __builtin_amdgcn_global_load_async_to_lds_b128( \
    (g_v4i*)(gp), (l_v4i*)(ldsp), 0, 0)
#else
#define CP16(ldsp, gp) (*(uint4*)(ldsp) = *(const uint4*)(gp))
#endif

__device__ __forceinline__ void wait_stage() {
#ifdef HAVE_ASYNC
#if __has_builtin(__builtin_amdgcn_s_wait_asynccnt)
  __builtin_amdgcn_s_wait_asynccnt(0);
#else
  asm volatile("s_wait_asynccnt 0x0" ::: "memory");
#endif
#endif
}

__device__ __forceinline__ float n2nf(float x) {
  if (x != x) return 0.f;
  if (x ==  INFINITY) return  1.f;
  if (x == -INFINITY) return -1.f;
  return x;
}

__device__ __forceinline__ v16bf frag2(const bf16* p0, const bf16* p1) {
  v8bf a = *(const v8bf*)p0;
  v8bf b = *(const v8bf*)p1;
  return __builtin_shufflevector(a, b, 0,1,2,3,4,5,6,7,8,9,10,11,12,13,14,15);
}

// ---------------------------------------------------------------------------
// WMMA GEMM: C[M,N] = act(A[M,K] @ W[N,K]^T + bias)
// A, W bf16 row-major. Block tile 256x64, K-step 32, 8 waves; wave w computes
// a 32x64 strip (2 M-frags x 4 N-frags = 8 WMMA per K-step, B frags reused).
// Double-buffered LDS staged with async global->LDS when available.
// M%256==0, N%64==0, K%32==0 (all shapes here satisfy this).
// ---------------------------------------------------------------------------
#define ASTR 40   // padded LDS row stride (bf16 elems): 80B, 16B-aligned
template<int RELU>
__global__ __launch_bounds__(256)
void gemm_wmma(const bf16* __restrict__ A, const bf16* __restrict__ W,
               const float* __restrict__ bias,
               float* __restrict__ Cf, bf16* __restrict__ Cb,
               int M, int Nn, int K)
{
  __shared__ __align__(16) bf16 As[2][256 * ASTR];
  __shared__ __align__(16) bf16 Bs[2][64 * ASTR];

  const int tid  = threadIdx.x;
  const int wave = tid >> 5;
  const int lane = tid & 31;
  const int lm   = lane & 15;
  const int hi   = lane >> 4;
  const int m0   = blockIdx.x * 256;
  const int n0   = blockIdx.y * 64;

  v8f acc[2][4] = {};

  // staging assignments: A -> thread t owns global row (m0+t), 4x16B chunks
  //                      B -> thread t owns (row = t>>2, col8 = (t&3)*8), 1x16B
  const bf16* Ag = A + (size_t)(m0 + tid) * K;
  bf16*       Al0 = &As[0][tid * ASTR];
  const int   brow = tid >> 2;
  const int   bcol = (tid & 3) << 3;
  const bf16* Wg = W + (size_t)(n0 + brow) * K + bcol;
  bf16*       Bl0 = &Bs[0][brow * ASTR + bcol];
  const size_t bufA = (size_t)(256 * ASTR);
  const size_t bufB = (size_t)(64 * ASTR);

  // stage tile 0
  {
    CP16(Al0 + 0,  Ag + 0);
    CP16(Al0 + 8,  Ag + 8);
    CP16(Al0 + 16, Ag + 16);
    CP16(Al0 + 24, Ag + 24);
    CP16(Bl0, Wg);
  }
  wait_stage();
  __syncthreads();

  int cur = 0;
  for (int k0 = 0; k0 < K; k0 += 32) {
    const bool more = (k0 + 32) < K;
    if (more) {
      if (k0 + 64 < K) {                   // global_prefetch_b8 for tile k+2
        __builtin_prefetch(Ag + k0 + 64, 0, 3);
        __builtin_prefetch(Wg + k0 + 64, 0, 3);
      }
      bf16* Al = Al0 + (cur ^ 1) * bufA;
      bf16* Bl = Bl0 + (cur ^ 1) * bufB;
      const bf16* Agk = Ag + k0 + 32;
      CP16(Al + 0,  Agk + 0);
      CP16(Al + 8,  Agk + 8);
      CP16(Al + 16, Agk + 16);
      CP16(Al + 24, Agk + 24);
      CP16(Bl, Wg + k0 + 32);
    }

    // A fragments (16x32): lanes 0-15: K 0-7|16-23 ; lanes 16-31: K 8-15|24-31
    const bf16* a0 = &As[cur][(wave * 32 + lm) * ASTR];
    const bf16* a1 = a0 + 16 * ASTR;
    v16bf af0 = frag2(a0 + 8 * hi, a0 + 16 + 8 * hi);
    v16bf af1 = frag2(a1 + 8 * hi, a1 + 16 + 8 * hi);

#pragma unroll
    for (int nt = 0; nt < 4; ++nt) {
      // B fragment (32x16): lane n = lm, lanes 0-15: K 0-7|16-23 ; lanes 16-31: K 8-15|24-31
      const bf16* br = &Bs[cur][(nt * 16 + lm) * ASTR];
      v16bf bfr = frag2(br + 8 * hi, br + 16 + 8 * hi);
      acc[0][nt] = __builtin_amdgcn_wmma_f32_16x16x32_bf16(
          false, af0, false, bfr, (short)0, acc[0][nt], false, false);
      acc[1][nt] = __builtin_amdgcn_wmma_f32_16x16x32_bf16(
          false, af1, false, bfr, (short)0, acc[1][nt], false, false);
    }

    if (more) {
      wait_stage();
      __syncthreads();
      cur ^= 1;
    }
  }

#pragma unroll
  for (int mf = 0; mf < 2; ++mf) {
#pragma unroll
    for (int nt = 0; nt < 4; ++nt) {
      int col = n0 + nt * 16 + lm;
      float bv = bias ? bias[col] : 0.f;
#pragma unroll
      for (int i = 0; i < 8; ++i) {
        int row = m0 + wave * 32 + mf * 16 + i + 8 * hi;  // C: VGPR i -> M = i + 8*hi
        float v = acc[mf][nt][i] + bv;
        if (RELU) v = fmaxf(v, 0.f);
        size_t o = (size_t)row * Nn + col;
        if (Cf) Cf[o] = v;
        if (Cb) Cb[o] = (bf16)v;
      }
    }
  }
}

// ---------------------------------------------------------------------------
__global__ __launch_bounds__(256)
void ln_kernel(const float* __restrict__ in, const float* __restrict__ g,
               const float* __restrict__ bt, float* outF, bf16* outB,
               int dim, int do_n2n)
{
  int row = blockIdx.x;
  const float* x = in + (size_t)row * dim;
  __shared__ float r1[256], r2[256];
  float s1 = 0.f, s2 = 0.f;
  for (int j = threadIdx.x; j < dim; j += 256) { float v = x[j]; s1 += v; s2 += v * v; }
  r1[threadIdx.x] = s1; r2[threadIdx.x] = s2; __syncthreads();
  for (int st = 128; st > 0; st >>= 1) {
    if (threadIdx.x < st) { r1[threadIdx.x] += r1[threadIdx.x + st]; r2[threadIdx.x] += r2[threadIdx.x + st]; }
    __syncthreads();
  }
  float mean = r1[0] / dim;
  float var  = r2[0] / dim - mean * mean;
  float rstd = rsqrtf(var + 1e-5f);
  for (int j = threadIdx.x; j < dim; j += 256) {
    float v = (x[j] - mean) * rstd * g[j] + bt[j];
    if (do_n2n) v = n2nf(v);
    if (outF) outF[(size_t)row * dim + j] = v;
    if (outB) outB[(size_t)row * dim + j] = (bf16)v;
  }
}

__global__ void cvt_bf16_kernel(const float* __restrict__ in, bf16* __restrict__ out, int n) {
  int i = blockIdx.x * 256 + threadIdx.x;
  if (i < n) out[i] = (bf16)in[i];
}

// gr = fourier(rel)@gw^T + gb, fused with aK = bf16(Kx+gr), aV = bf16(Vx+gr)
__global__ __launch_bounds__(256)
void gr_kernel(const float* __restrict__ coords, const float* __restrict__ Sp,
               const float* __restrict__ Ss, const float* __restrict__ gw,
               const float* __restrict__ gb, const float* __restrict__ Kx,
               const float* __restrict__ Vx, bf16* __restrict__ aK, bf16* __restrict__ aV)
{
  int r  = blockIdx.x;
  int n  = r & (NB - 1);
  int bs = r >> 11;          // b*8+s
  int b  = bs >> 3;
  __shared__ float feat[40];
  int tid = threadIdx.x;
  if (tid < 39) {
    float rel[3];
#pragma unroll
    for (int j = 0; j < 3; ++j) {
      float ss = Ss[bs * 3 + j];
      ss = fminf(fmaxf(ss, 0.3f), 5.0f) + 0.1f;
      float dd = (coords[((size_t)b * NB + n) * 3 + j] - Sp[bs * 3 + j]) / (ss * 5.0f + 1e-4f);
      rel[j] = fminf(fmaxf(dd, -3.f), 3.f);
    }
    if (tid < 3) feat[tid] = rel[tid];
    else {
      int idx = tid - 3; int c = idx / 12; int rem = idx - c * 12; int l = rem >> 1;
      float sc = rel[c] * (float)(1 << l) * 6.28318530717958647692f;
      feat[tid] = (rem & 1) ? __cosf(sc) : __sinf(sc);
    }
  }
  __syncthreads();
  int d = tid;
  const float* grow = gw + d * 39;
  float acc = gb[d];
#pragma unroll
  for (int i = 0; i < 39; ++i) acc += feat[i] * grow[i];
  size_t bnd = ((size_t)b * NB + n) * DSN + d;
  size_t rd  = (size_t)r * DSN + d;
  aK[rd] = (bf16)(Kx[bnd] + acc);
  aV[rd] = (bf16)(Vx[bnd] + acc);
}

__global__ __launch_bounds__(256)
void dots_kernel(const float* __restrict__ kbuf, const float* __restrict__ q,
                 float* __restrict__ dots)
{
  int r    = blockIdx.x * 8 + (threadIdx.x >> 5);
  int lane = threadIdx.x & 31;
  int bs   = r >> 11;
  const float* kr = kbuf + (size_t)r * DSN;
  const float* qr = q + bs * DSN;
  float acc = 0.f;
  for (int j = lane; j < DSN; j += 32) acc += kr[j] * qr[j];
  for (int off = 16; off > 0; off >>= 1) acc += __shfl_xor(acc, off, 32);
  if (lane == 0) dots[r] = fminf(fmaxf(acc * 0.0625f, -30.f), 30.f);
}

__global__ void softmax_s(const float* __restrict__ dots, float* __restrict__ attn) {
  int idx = blockIdx.x * 256 + threadIdx.x;
  if (idx >= BB * NB) return;
  int b = idx / NB, n = idx % NB;
  float v[SLOTS]; float mx = -1e30f;
#pragma unroll
  for (int s = 0; s < SLOTS; ++s) { v[s] = dots[((size_t)(b * SLOTS + s)) * NB + n]; mx = fmaxf(mx, v[s]); }
  float sum = 0.f;
#pragma unroll
  for (int s = 0; s < SLOTS; ++s) { v[s] = __expf(v[s] - mx); sum += v[s]; }
  float inv = 1.f / sum;
#pragma unroll
  for (int s = 0; s < SLOTS; ++s) attn[((size_t)(b * SLOTS + s)) * NB + n] = v[s] * inv + 1e-8f;
}

__global__ __launch_bounds__(256)
void attn_norm(float* __restrict__ attn) {
  int bs = blockIdx.x;
  float* a = attn + (size_t)bs * NB;
  __shared__ float red[256];
  float s = 0.f;
  for (int j = threadIdx.x; j < NB; j += 256) s += a[j];
  red[threadIdx.x] = s; __syncthreads();
  for (int st = 128; st > 0; st >>= 1) {
    if (threadIdx.x < st) red[threadIdx.x] += red[threadIdx.x + st];
    __syncthreads();
  }
  float inv = 1.f / (red[0] + 1e-8f);
  for (int j = threadIdx.x; j < NB; j += 256) a[j] *= inv;
}

__global__ __launch_bounds__(256)
void update_kernel(const float* __restrict__ attn, const float* __restrict__ coords,
                   const float* __restrict__ vbuf, float* __restrict__ Sp,
                   float* __restrict__ Ss, float* __restrict__ upd)
{
  int bs = blockIdx.x; int b = bs >> 3;
  const float* a = attn + (size_t)bs * NB;
  __shared__ float spl[3];
  int wave = threadIdx.x >> 5, lane = threadIdx.x & 31;
  if (wave < 3) {
    float acc = 0.f;
    for (int n = lane; n < NB; n += 32) acc += coords[((size_t)b * NB + n) * 3 + wave] * a[n];
    for (int off = 16; off > 0; off >>= 1) acc += __shfl_xor(acc, off, 32);
    if (!lane) { spl[wave] = acc; Sp[bs * 3 + wave] = acc; }
  }
  __syncthreads();
  if (wave < 3) {
    float p = spl[wave]; float acc = 0.f;
    for (int n = lane; n < NB; n += 32) {
      float dd = coords[((size_t)b * NB + n) * 3 + wave] - p;
      acc += dd * dd * a[n];
    }
    for (int off = 16; off > 0; off >>= 1) acc += __shfl_xor(acc, off, 32);
    if (!lane) Ss[bs * 3 + wave] = fminf(fmaxf(sqrtf(acc + 1e-6f), 0.2f), 5.0f);
  }
  int d = threadIdx.x;
  const float* v = vbuf + (size_t)bs * NB * DSN;
  float acc = 0.f;
  for (int n = 0; n < NB; ++n) acc += v[(size_t)n * DSN + d] * a[n];
  upd[bs * DSN + d] = n2nf(acc);
}

__global__ void smallmm(const float* __restrict__ A, const float* __restrict__ W,
                        const float* __restrict__ bias, float* __restrict__ C,
                        int M, int Nout, int K, int relu)
{
  int idx = blockIdx.x * 256 + threadIdx.x;
  if (idx >= M * Nout) return;
  int m = idx / Nout, o = idx % Nout;
  float acc = bias ? bias[o] : 0.f;
  const float* a = A + (size_t)m * K;
  const float* w = W + (size_t)o * K;
  for (int k = 0; k < K; ++k) acc += a[k] * w[k];
  if (relu) acc = fmaxf(acc, 0.f);
  C[idx] = acc;
}

__global__ void gru_gate(const float* __restrict__ gi, const float* __restrict__ gh,
                         float* __restrict__ slots)
{
  int r = blockIdx.x, d = threadIdx.x;
  const float* gir = gi + r * 768;
  const float* ghr = gh + r * 768;
  float rg  = 1.f / (1.f + __expf(-(gir[d]       + ghr[d])));
  float zg  = 1.f / (1.f + __expf(-(gir[256 + d] + ghr[256 + d])));
  float ngv = tanhf(gir[512 + d] + rg * ghr[512 + d]);
  float h   = slots[r * DSN + d];
  slots[r * DSN + d] = (1.f - zg) * ngv + zg * h;
}

__global__ void add_n2n(float* __restrict__ slots, const float* __restrict__ m2) {
  int i = blockIdx.x * 256 + threadIdx.x;
  slots[i] = n2nf(slots[i] + m2[i]);
}

__global__ void init_kernel(const float* __restrict__ slots_p, const float* __restrict__ Sp0,
                            const float* __restrict__ Ss0, float* __restrict__ slots,
                            float* __restrict__ Sp, float* __restrict__ Ss)
{
  int idx = blockIdx.x * 256 + threadIdx.x;
  if (idx < BB * SLOTS * DSN) slots[idx] = slots_p[idx & (SLOTS * DSN - 1)];
  if (idx < BB * SLOTS * 3) { int sj = idx % (SLOTS * 3); Sp[idx] = Sp0[sj]; Ss[idx] = Ss0[sj]; }
}

// ---------------------------------------------------------------------------
extern "C" void kernel_launch(void* const* d_in, const int* in_sizes, int n_in,
                              void* d_out, int out_size, void* d_ws, size_t ws_size,
                              hipStream_t stream)
{
  (void)in_sizes; (void)n_in; (void)out_size; (void)ws_size;
  const float* inputs = (const float*)d_in[0];
  const float* coords = (const float*)d_in[1];
  const float* slots_p= (const float*)d_in[2];
  const float* Ss0    = (const float*)d_in[3];
  const float* Sp0    = (const float*)d_in[4];
  const float* Qw     = (const float*)d_in[5];
  const float* Kw     = (const float*)d_in[6];
  const float* Vw     = (const float*)d_in[7];
  const float* gw     = (const float*)d_in[8];
  const float* gb     = (const float*)d_in[9];
  const float* f1w    = (const float*)d_in[10];
  const float* f1b    = (const float*)d_in[11];
  const float* f2w    = (const float*)d_in[12];
  const float* f2b    = (const float*)d_in[13];
  const float* ng     = (const float*)d_in[14];
  const float* nbv    = (const float*)d_in[15];
  const float* wih    = (const float*)d_in[16];
  const float* whh    = (const float*)d_in[17];
  const float* bih    = (const float*)d_in[18];
  const float* bhh    = (const float*)d_in[19];
  const float* m_ng   = (const float*)d_in[20];
  const float* m_nb   = (const float*)d_in[21];
  const float* m1w    = (const float*)d_in[22];
  const float* m1b    = (const float*)d_in[23];
  const float* m2w    = (const float*)d_in[24];
  const float* m2b    = (const float*)d_in[25];
  const float* i0g    = (const float*)d_in[26];
  const float* i0b    = (const float*)d_in[27];
  const float* i1w    = (const float*)d_in[28];
  const float* i1b    = (const float*)d_in[29];
  const float* i2w    = (const float*)d_in[30];
  const float* i2b    = (const float*)d_in[31];
  const float* i3g    = (const float*)d_in[32];
  const float* i3b    = (const float*)d_in[33];
  const float* fw     = (const float*)d_in[34];
  const float* fb     = (const float*)d_in[35];

  char* base = (char*)d_ws;
  size_t off = 0;
  auto alloc = [&](size_t bytes) -> char* {
    char* p = base + off; off += (bytes + 255) & ~(size_t)255; return p;
  };

  bf16* i1w_bf = (bf16*)alloc((size_t)768 * 768 * 2);
  bf16* i2w_bf = (bf16*)alloc((size_t)256 * 768 * 2);
  bf16* Kw_bf  = (bf16*)alloc((size_t)256 * 256 * 2);
  bf16* Vw_bf  = (bf16*)alloc((size_t)256 * 256 * 2);
  bf16* f1w_bf = (bf16*)alloc((size_t)256 * 256 * 2);
  bf16* f2w_bf = (bf16*)alloc((size_t)256 * 256 * 2);
  float* Kx    = (float*)alloc((size_t)MENC * DSN * 4);
  float* Vx    = (float*)alloc((size_t)MENC * DSN * 4);
  float* slots = (float*)alloc(32 * 256 * 4);
  float* snb   = (float*)alloc(32 * 256 * 4);
  float* qb    = (float*)alloc(32 * 256 * 4);
  float* Spb   = (float*)alloc(32 * 3 * 4);
  float* Ssb   = (float*)alloc(32 * 3 * 4);
  float* dots  = (float*)alloc((size_t)RTOT * 4);
  float* upd   = (float*)alloc(32 * 256 * 4);
  float* gi    = (float*)alloc(32 * 768 * 4);
  float* gh    = (float*)alloc(32 * 768 * 4);
  float* hb    = (float*)alloc(32 * 256 * 4);
  float* m1    = (float*)alloc(32 * 1024 * 4);
  float* m2    = (float*)alloc(32 * 256 * 4);

  // big union region: loop view = aK + aV + tmpbf (bf16) + kv (f32)
  size_t fr = (size_t)RTOT * DSN;             // 16.7M elements
  char* big = alloc(fr * 2 * 3 + fr * 4);
  // encoder view (used only before the loop; fits inside the union)
  bf16* xin_bf = (bf16*)big;
  bf16* h1_bf  = (bf16*)(big + (size_t)MENC * 768 * 2);
  float* xpre  = (float*)(big + (size_t)MENC * 768 * 4);
  bf16* xbf    = (bf16*)(big + (size_t)MENC * 768 * 4 + (size_t)MENC * 256 * 4);
  // loop view
  bf16* aK    = (bf16*)big;
  bf16* aV    = aK + fr;
  bf16* tmpbf = aV + fr;
  float* kv   = (float*)(big + fr * 2 * 3);

  float* out_slots = (float*)d_out;
  float* attn      = (float*)d_out + 32 * 256;   // (B,S,N) region of output

  // --- weight conversion to bf16 ---
  cvt_bf16_kernel<<<(768 * 768 + 255) / 256, 256, 0, stream>>>(i1w, i1w_bf, 768 * 768);
  cvt_bf16_kernel<<<(256 * 768 + 255) / 256, 256, 0, stream>>>(i2w, i2w_bf, 256 * 768);
  cvt_bf16_kernel<<<(256 * 256 + 255) / 256, 256, 0, stream>>>(Kw, Kw_bf, 256 * 256);
  cvt_bf16_kernel<<<(256 * 256 + 255) / 256, 256, 0, stream>>>(Vw, Vw_bf, 256 * 256);
  cvt_bf16_kernel<<<(256 * 256 + 255) / 256, 256, 0, stream>>>(f1w, f1w_bf, 256 * 256);
  cvt_bf16_kernel<<<(256 * 256 + 255) / 256, 256, 0, stream>>>(f2w, f2w_bf, 256 * 256);

  // --- encoder ---
  ln_kernel<<<MENC, 256, 0, stream>>>(inputs, i0g, i0b, nullptr, xin_bf, DIN, 0);
  gemm_wmma<1><<<dim3(MENC / 256, 768 / 64), 256, 0, stream>>>(
      xin_bf, i1w_bf, i1b, nullptr, h1_bf, MENC, 768, 768);
  gemm_wmma<0><<<dim3(MENC / 256, 256 / 64), 256, 0, stream>>>(
      h1_bf, i2w_bf, i2b, xpre, nullptr, MENC, 256, 768);
  ln_kernel<<<MENC, 256, 0, stream>>>(xpre, i3g, i3b, nullptr, xbf, DSN, 1);
  gemm_wmma<0><<<dim3(MENC / 256, 4), 256, 0, stream>>>(
      xbf, Kw_bf, nullptr, Kx, nullptr, MENC, 256, 256);
  gemm_wmma<0><<<dim3(MENC / 256, 4), 256, 0, stream>>>(
      xbf, Vw_bf, nullptr, Vx, nullptr, MENC, 256, 256);

  init_kernel<<<32, 256, 0, stream>>>(slots_p, Sp0, Ss0, slots, Spb, Ssb);

  // --- slot attention iterations (t = 0..ITERS; stop_gradient is identity fwd) ---
  for (int t = 0; t <= 3; ++t) {
    ln_kernel<<<32, 256, 0, stream>>>(slots, ng, nbv, snb, nullptr, DSN, 0);
    smallmm<<<(32 * 256 + 255) / 256, 256, 0, stream>>>(snb, Qw, nullptr, qb, 32, 256, 256, 0);
    gr_kernel<<<RTOT, 256, 0, stream>>>(coords, Spb, Ssb, gw, gb, Kx, Vx, aK, aV);

    // k = f_mlp(Kx + gr)
    gemm_wmma<1><<<dim3(RTOT / 256, 4), 256, 0, stream>>>(
        aK, f1w_bf, f1b, nullptr, tmpbf, RTOT, 256, 256);
    gemm_wmma<0><<<dim3(RTOT / 256, 4), 256, 0, stream>>>(
        tmpbf, f2w_bf, f2b, kv, nullptr, RTOT, 256, 256);

    dots_kernel<<<RTOT / 8, 256, 0, stream>>>(kv, qb, dots);
    softmax_s<<<(BB * NB + 255) / 256, 256, 0, stream>>>(dots, attn);
    attn_norm<<<32, 256, 0, stream>>>(attn);

    if (t < 3) {
      // v = f_mlp(Vx + gr)  (kv buffer reused; dots already consumed)
      gemm_wmma<1><<<dim3(RTOT / 256, 4), 256, 0, stream>>>(
          aV, f1w_bf, f1b, nullptr, tmpbf, RTOT, 256, 256);
      gemm_wmma<0><<<dim3(RTOT / 256, 4), 256, 0, stream>>>(
          tmpbf, f2w_bf, f2b, kv, nullptr, RTOT, 256, 256);

      update_kernel<<<32, 256, 0, stream>>>(attn, coords, kv, Spb, Ssb, upd);

      // GRU
      smallmm<<<(32 * 768 + 255) / 256, 256, 0, stream>>>(upd, wih, bih, gi, 32, 768, 256, 0);
      smallmm<<<(32 * 768 + 255) / 256, 256, 0, stream>>>(slots, whh, bhh, gh, 32, 768, 256, 0);
      gru_gate<<<32, 256, 0, stream>>>(gi, gh, slots);

      // residual slot MLP
      ln_kernel<<<32, 256, 0, stream>>>(slots, m_ng, m_nb, hb, nullptr, DSN, 0);
      smallmm<<<(32 * 1024 + 255) / 256, 256, 0, stream>>>(hb, m1w, m1b, m1, 32, 1024, 256, 1);
      smallmm<<<(32 * 256 + 255) / 256, 256, 0, stream>>>(m1, m2w, m2b, m2, 32, 256, 1024, 0);
      add_n2n<<<32, 256, 0, stream>>>(slots, m2);
    }
  }

  // out_slots = slots_prev @ fw^T + fb  (slots unchanged during t==ITERS pass)
  smallmm<<<(32 * 256 + 255) / 256, 256, 0, stream>>>(slots, fw, fb, out_slots, 32, 256, 256, 0);
}
